// UNetEncoderWithMamba_24017457119251
// MI455X (gfx1250) — matrix operand order repaired and checked by reading, not compiled, not run on hardware
//
#include <hip/hip_runtime.h>
#include <hip/hip_bf16.h>

// ---------------------------------------------------------------------------
// UNet encoder block with 3-direction Mamba (CDNA5 / gfx1250, wave32, WMMA)
// B=2, C=96, D=H=W=20 -> L=8000, BL=16000
// D_INNER=192, D_STATE=16, D_CONV=4, DT_RANK=6
// Selective scan parallelized as a 3-phase chunked scan (100 chunks x 80).
// ---------------------------------------------------------------------------

#define LSEQ   8000
#define BATCH  2
#define BL     (BATCH * LSEQ)
#define CCH    96
#define DIN    192
#define DSTATE 16
#define DTRANK 6
#define XDBL_LD 48   // padded stride for (DT_RANK + 2*D_STATE = 38) columns
#define NCH    100   // scan chunks
#define CLEN   80    // chunk length (NCH * CLEN == LSEQ)

typedef _Float16 half_t;
typedef __attribute__((ext_vector_type(16))) _Float16 v16h;
typedef __attribute__((ext_vector_type(8)))  _Float16 v8h;
typedef __attribute__((ext_vector_type(8)))  float    v8f;

__device__ __forceinline__ float silu_f(float x)  { return x / (1.0f + __expf(-x)); }
__device__ __forceinline__ float gelu_f(float x)  { return 0.5f * x * (1.0f + erff(x * 0.70710678118654752f)); }
__device__ __forceinline__ float softplus_f(float x){ return (x > 20.0f) ? x : log1pf(__expf(x)); }

// ---------------------------------------------------------------------------
// fp32 -> f16 conversion (weights)
// ---------------------------------------------------------------------------
__global__ void cvt_f32_f16_kernel(const float* __restrict__ src,
                                   half_t* __restrict__ dst, int n) {
  int t = blockIdx.x * blockDim.x + threadIdx.x;
  if (t < n) dst[t] = (half_t)src[t];
}

// ---------------------------------------------------------------------------
// Fused channel-first LN (eps 1e-6, ln_w/ln_b) + second LN (eps 1e-5,
// mln_w/mln_b). Both normalize the 96-channel vector at each position, so the
// per-direction _ln_last collapses into this single pre-pass.
// ---------------------------------------------------------------------------
__global__ void ln_cf_mln_kernel(const float* __restrict__ x,
                                 const float* __restrict__ lw, const float* __restrict__ lb,
                                 const float* __restrict__ mw, const float* __restrict__ mb,
                                 float* __restrict__ x2) {
  int t = blockIdx.x * blockDim.x + threadIdx.x;
  if (t >= BL) return;
  int b = t / LSEQ, pos = t % LSEQ;
  const float* xp = x + (size_t)b * CCH * LSEQ + pos;
  float s = 0.f;
  for (int c = 0; c < CCH; ++c) s += xp[(size_t)c * LSEQ];
  float u = s * (1.0f / CCH);
  float v = 0.f;
  for (int c = 0; c < CCH; ++c) { float d = xp[(size_t)c * LSEQ] - u; v += d * d; }
  float inv1 = rsqrtf(v * (1.0f / CCH) + 1e-6f);
  float s2 = 0.f, q2 = 0.f;
  for (int c = 0; c < CCH; ++c) {
    float y = lw[c] * ((xp[(size_t)c * LSEQ] - u) * inv1) + lb[c];
    s2 += y; q2 += y * y;
  }
  float u2 = s2 * (1.0f / CCH);
  float inv2 = rsqrtf(fmaxf(q2 * (1.0f / CCH) - u2 * u2, 0.f) + 1e-5f);
  float* op = x2 + (size_t)b * CCH * LSEQ + pos;
  for (int c = 0; c < CCH; ++c) {
    float y = lw[c] * ((xp[(size_t)c * LSEQ] - u) * inv1) + lb[c];
    op[(size_t)c * LSEQ] = mw[c] * ((y - u2) * inv2) + mb[c];
  }
}

// ---------------------------------------------------------------------------
// Gather permuted scan-order sequence, channel-last f16: xf[b,l,c]
// p=0: l=(d,h,w) ; p=1: l=(h,w,d) ; p=2: l=(w,d,h) over the original grid.
// ---------------------------------------------------------------------------
__global__ void gather_kernel(const float* __restrict__ x2, half_t* __restrict__ xfh, int p) {
  int t = blockIdx.x * blockDim.x + threadIdx.x;
  if (t >= BL * CCH) return;
  int c = t % CCH;
  int l = (t / CCH) % LSEQ;
  int b = t / (CCH * LSEQ);
  int src;
  if (p == 0) {
    src = l;
  } else if (p == 1) {
    int hh = l / 400, ww = (l / 20) % 20, dd = l % 20;
    src = dd * 400 + hh * 20 + ww;
  } else {
    int ww = l / 400, dd = (l / 20) % 20, hh = l % 20;
    src = dd * 400 + hh * 20 + ww;
  }
  xfh[t] = (half_t)x2[((size_t)b * CCH + c) * LSEQ + src];
}

// ---------------------------------------------------------------------------
// Generic WMMA f16 GEMM:  Out[M x N] = A[M x K] * W[N x K]^T  (+bias, act, res)
// One wave per 16x16 tile; K-loop of v_wmma_f32_16x16x32_f16.
// Operand layouts used here (contiguous k, same order in A and B):
//   A frag: lane m(0..15) elems e -> K(kk+e) (contig 16 f16);
//           lane m+16 -> K offset +16.
//   B frag: lane n(0..15) elems e -> B[kk+e][n] = W[n][kk+e] (contig 16 f16);
//           lanes 16..31 -> K offset +16.
//   C/D  : VGPR j -> M = tile_m + 8*half + j, N = tile_n + (lane&15).
// out_mode: 0=f32 row-major, 1=f16 row-major, 2=f32 channel-first + res_cl.
// ---------------------------------------------------------------------------
__global__ void gemm_wmma_kernel(const half_t* __restrict__ A,
                                 const half_t* __restrict__ W,
                                 void* __restrict__ outp,
                                 const float* __restrict__ bias,
                                 const float* __restrict__ res_cf,
                                 const float* __restrict__ res_cl,
                                 int M, int Ntiles, int K,
                                 int lda, int ldb, int ldo,
                                 int n_valid, int act, int out_mode) {
  int wave = (blockIdx.x * blockDim.x + threadIdx.x) >> 5;
  int total = (M >> 4) * Ntiles;
  if (wave >= total) return;                       // wave-uniform (EXEC all-1 for WMMA)
  int tm = wave / Ntiles;
  int tn = wave % Ntiles;
  int lane = threadIdx.x & 31;
  int lr = lane & 15;
  int hf = lane >> 4;

  int arow = tm * 16 + lr;
  int wrow = tn * 16 + lr;
  int wrow_c = (wrow < n_valid) ? wrow : (n_valid - 1); // clamp OOB weight rows
  const half_t* aptr = A + (size_t)arow * lda + hf * 16;
  const half_t* wptr = W + (size_t)wrow_c * ldb + hf * 16;

  v8f acc = {};
  for (int kk = 0; kk < K; kk += 32) {
    v16h af = *(const v16h*)(aptr + kk);
    v16h bf = *(const v16h*)(wptr + kk);
    acc = __builtin_amdgcn_wmma_f32_16x16x32_f16(false, af, false, bf,
                                                 (short)0, acc, false, false);
  }

  int col = tn * 16 + lr;
  if (col < n_valid) {
    int mbase = tm * 16 + hf * 8;
#pragma unroll
    for (int j = 0; j < 8; ++j) {
      int m = mbase + j;
      float v = acc[j];
      if (bias)     v += bias[col];
      if (act == 1) v = gelu_f(v);
      int b = m / LSEQ, l = m % LSEQ;
      if (res_cf)   v += res_cf[((size_t)(b * CCH + col)) * LSEQ + l];
      if (out_mode == 0) {
        ((float*)outp)[(size_t)m * ldo + col] = v;
      } else if (out_mode == 1) {
        ((half_t*)outp)[(size_t)m * ldo + col] = (half_t)v;
      } else {
        v += res_cl[(size_t)m * CCH + col];
        ((float*)outp)[((size_t)(b * CCH + col)) * LSEQ + l] = v;
      }
    }
  }
}

// ---------------------------------------------------------------------------
// Depthwise causal conv (width 4) + SiLU over scan order; input = xz[:, :192].
// ---------------------------------------------------------------------------
__global__ void conv_silu_kernel(const float* __restrict__ xz,
                                 const float* __restrict__ cw,
                                 const float* __restrict__ cb,
                                 float* __restrict__ xc, half_t* __restrict__ xch) {
  int t = blockIdx.x * blockDim.x + threadIdx.x;
  if (t >= BL * DIN) return;
  int d = t % DIN;
  int l = (t / DIN) % LSEQ;
  int b = t / (DIN * LSEQ);
  float acc = cb[d];
#pragma unroll
  for (int j = 0; j < 4; ++j) {
    int ll = l + j - 3;
    if (ll >= 0) acc += cw[d * 4 + j] * xz[((size_t)b * LSEQ + ll) * (2 * DIN) + d];
  }
  float v = silu_f(acc);
  size_t o = ((size_t)b * LSEQ + l) * DIN + d;
  xc[o]  = v;
  xch[o] = (half_t)v;
}

// ---------------------------------------------------------------------------
// dt = softplus(dt_r @ Wdt^T + bdt); dt_r = x_dbl[:, 0:6]
// ---------------------------------------------------------------------------
__global__ void dt_kernel(const float* __restrict__ xdbl,
                          const float* __restrict__ Wdt, const float* __restrict__ bdt,
                          float* __restrict__ dt) {
  int t = blockIdx.x * blockDim.x + threadIdx.x;
  if (t >= BL * DIN) return;
  int d = t % DIN;
  size_t row = t / DIN;
  float acc = bdt[d];
#pragma unroll
  for (int r = 0; r < DTRANK; ++r) acc += xdbl[row * XDBL_LD + r] * Wdt[d * DTRANK + r];
  dt[row * DIN + d] = softplus_f(acc);
}

// ---------------------------------------------------------------------------
// Chunked selective scan.
//   recurrence per (b,d,s): h = exp(dt*A[s])*h + dt*xc*B[s]
// Phase 1: per-chunk local carries (prod of a, h_end with h0=0).
//   2 chunk-sequences per wave (16 state lanes each); 19200 waves.
// ---------------------------------------------------------------------------
__global__ void scan_p1_kernel(const float* __restrict__ dt,
                               const float* __restrict__ xc,
                               const float* __restrict__ xdbl,
                               const float* __restrict__ A_log,
                               float* __restrict__ carryA,
                               float* __restrict__ carryH) {
  int wave = (blockIdx.x * blockDim.x + threadIdx.x) >> 5;
  if (wave >= (BATCH * DIN * NCH) / 2) return;     // uniform exit
  int lane = threadIdx.x & 31;
  int lr = lane & 15, hf = lane >> 4;
  int cs = wave * 2 + hf;                          // 0 .. 38399
  int chunk = cs % NCH;
  int bd = cs / NCH;
  int b = bd / DIN, d = bd % DIN;
  float A = -__expf(A_log[d * DSTATE + lr]);
  float ap = 1.f, h = 0.f;
  int l0 = chunk * CLEN;
  for (int i = 0; i < CLEN; ++i) {
    size_t row = (size_t)b * LSEQ + l0 + i;
    float dtv = dt[row * DIN + d];
    float xcv = xc[row * DIN + d];
    float Bm  = xdbl[row * XDBL_LD + DTRANK + lr];
    float a   = __expf(dtv * A);
    ap *= a;
    h = a * h + (dtv * xcv) * Bm;
  }
  size_t ci = ((size_t)bd * NCH + chunk) * DSTATE + lr;
  carryA[ci] = ap;
  carryH[ci] = h;
}

// ---------------------------------------------------------------------------
// Phase 2: serial scan over chunk carries -> incoming state per chunk.
// One thread per (b,d,s): 6144 threads, 100 steps.
// ---------------------------------------------------------------------------
__global__ void scan_p2_kernel(const float* __restrict__ carryA,
                               const float* __restrict__ carryH,
                               float* __restrict__ hinit) {
  int t = blockIdx.x * blockDim.x + threadIdx.x;
  if (t >= BATCH * DIN * DSTATE) return;
  int s = t % DSTATE;
  int bd = t / DSTATE;
  float hi = 0.f;
  for (int k = 0; k < NCH; ++k) {
    size_t ci = ((size_t)bd * NCH + k) * DSTATE + s;
    hinit[ci] = hi;
    hi = carryA[ci] * hi + carryH[ci];
  }
}

// ---------------------------------------------------------------------------
// Phase 3: replay each chunk with correct incoming state; reduce over the 16
// state lanes, add D-skip, gate with silu(z), store f16 rows for Wout GEMM.
// ---------------------------------------------------------------------------
__global__ void scan_p3_kernel(const float* __restrict__ dt,
                               const float* __restrict__ xc,
                               const float* __restrict__ xdbl,
                               const float* __restrict__ xz,
                               const float* __restrict__ A_log,
                               const float* __restrict__ Dp,
                               const float* __restrict__ hinit,
                               half_t* __restrict__ ygh) {
  int wave = (blockIdx.x * blockDim.x + threadIdx.x) >> 5;
  if (wave >= (BATCH * DIN * NCH) / 2) return;     // uniform exit
  int lane = threadIdx.x & 31;
  int lr = lane & 15, hf = lane >> 4;
  int cs = wave * 2 + hf;
  int chunk = cs % NCH;
  int bd = cs / NCH;
  int b = bd / DIN, d = bd % DIN;
  float A   = -__expf(A_log[d * DSTATE + lr]);
  float Dpv = Dp[d];
  float h   = hinit[((size_t)bd * NCH + chunk) * DSTATE + lr];
  int l0 = chunk * CLEN;
  for (int i = 0; i < CLEN; ++i) {
    size_t row = (size_t)b * LSEQ + l0 + i;
    float dtv = dt[row * DIN + d];
    float xcv = xc[row * DIN + d];
    float Bm  = xdbl[row * XDBL_LD + DTRANK + lr];
    float Cm  = xdbl[row * XDBL_LD + DTRANK + DSTATE + lr];
    float a   = __expf(dtv * A);
    h = a * h + (dtv * xcv) * Bm;
    float part = h * Cm;
    part += __shfl_xor(part, 1, 32);
    part += __shfl_xor(part, 2, 32);
    part += __shfl_xor(part, 4, 32);
    part += __shfl_xor(part, 8, 32);
    if (lr == 0) {
      float y  = part + xcv * Dpv;
      float zv = xz[row * (2 * DIN) + DIN + d];
      ygh[row * DIN + d] = (half_t)(y * silu_f(zv));
    }
  }
}

// ---------------------------------------------------------------------------
// Channel-last LN (eps 1e-6, ln_w/ln_b) -> f16 rows for fc1.
// ---------------------------------------------------------------------------
__global__ void ln_cl_kernel(const float* __restrict__ in,
                             const float* __restrict__ lw, const float* __restrict__ lb,
                             half_t* __restrict__ outh) {
  int t = blockIdx.x * blockDim.x + threadIdx.x;
  if (t >= BL) return;
  const float* p = in + (size_t)t * CCH;
  float s = 0.f;
  for (int c = 0; c < CCH; ++c) s += p[c];
  float u = s * (1.0f / CCH);
  float v = 0.f;
  for (int c = 0; c < CCH; ++c) { float d = p[c] - u; v += d * d; }
  float inv = rsqrtf(v * (1.0f / CCH) + 1e-6f);
  half_t* o = outh + (size_t)t * CCH;
  for (int c = 0; c < CCH; ++c) o[c] = (half_t)(lw[c] * ((p[c] - u) * inv) + lb[c]);
}

// ---------------------------------------------------------------------------
extern "C" void kernel_launch(void* const* d_in, const int* in_sizes, int n_in,
                              void* d_out, int out_size, void* d_ws, size_t ws_size,
                              hipStream_t stream) {
  const float* x      = (const float*)d_in[0];
  const float* ln_w   = (const float*)d_in[1];
  const float* ln_b   = (const float*)d_in[2];
  const float* mln_w  = (const float*)d_in[3];
  const float* mln_b  = (const float*)d_in[4];
  const float* Win    = (const float*)d_in[5];
  const float* conv_w = (const float*)d_in[6];
  const float* conv_b = (const float*)d_in[7];
  const float* Wx     = (const float*)d_in[8];
  const float* Wdt    = (const float*)d_in[9];
  const float* bdt    = (const float*)d_in[10];
  const float* A_log  = (const float*)d_in[11];
  const float* Dp     = (const float*)d_in[12];
  const float* Wout   = (const float*)d_in[13];
  const float* proj_w = (const float*)d_in[14];
  const float* proj_b = (const float*)d_in[15];
  const float* fc1_w  = (const float*)d_in[16];
  const float* fc1_b  = (const float*)d_in[17];
  const float* fc2_w  = (const float*)d_in[18];
  const float* fc2_b  = (const float*)d_in[19];

  size_t off = 0;
  auto carve = [&](size_t bytes) -> void* {
    void* p = (char*)d_ws + off;
    off += (bytes + 255) & ~(size_t)255;
    return p;
  };
  float*  x2    = (float*) carve((size_t)BL * CCH * 4);
  half_t* xfh   = (half_t*)carve((size_t)BL * CCH * 2);
  float*  xz    = (float*) carve((size_t)BL * 2 * DIN * 4);
  float*  xc    = (float*) carve((size_t)BL * DIN * 4);
  half_t* xch   = (half_t*)carve((size_t)BL * DIN * 2);
  float*  xdbl  = (float*) carve((size_t)BL * XDBL_LD * 4);
  float*  dtb   = (float*) carve((size_t)BL * DIN * 4);
  half_t* ygh   = (half_t*)carve((size_t)BL * DIN * 2);
  half_t* cath  = (half_t*)carve((size_t)BL * 3 * CCH * 2);
  float*  carA  = (float*) carve((size_t)BATCH * DIN * NCH * DSTATE * 4);
  float*  carH  = (float*) carve((size_t)BATCH * DIN * NCH * DSTATE * 4);
  float*  hini  = (float*) carve((size_t)BATCH * DIN * NCH * DSTATE * 4);
  half_t* Winh  = (half_t*)carve((size_t)2 * DIN * CCH * 2);
  half_t* Wxh   = (half_t*)carve((size_t)(DTRANK + 2 * DSTATE) * DIN * 2);
  half_t* Wouth = (half_t*)carve((size_t)CCH * DIN * 2);
  half_t* projh = (half_t*)carve((size_t)CCH * 3 * CCH * 2);
  half_t* fc1h  = (half_t*)carve((size_t)4 * CCH * CCH * 2);
  half_t* fc2h  = (half_t*)carve((size_t)CCH * 4 * CCH * 2);
  // post-stage aliases (per-direction buffers are dead by then)
  float*  ores = xc;                 // (BL x 96) f32
  half_t* hh   = xch;                // (BL x 96) f16
  half_t* gh   = (half_t*)xz;        // (BL x 384) f16

  auto cdiv = [](int a, int b) { return (a + b - 1) / b; };
  const int TPB = 256;

  // weights -> f16
  cvt_f32_f16_kernel<<<cdiv(2*DIN*CCH,TPB),TPB,0,stream>>>(Win,  Winh,  2*DIN*CCH);
  cvt_f32_f16_kernel<<<cdiv((DTRANK+2*DSTATE)*DIN,TPB),TPB,0,stream>>>(Wx, Wxh, (DTRANK+2*DSTATE)*DIN);
  cvt_f32_f16_kernel<<<cdiv(CCH*DIN,TPB),TPB,0,stream>>>(Wout, Wouth, CCH*DIN);
  cvt_f32_f16_kernel<<<cdiv(CCH*3*CCH,TPB),TPB,0,stream>>>(proj_w, projh, CCH*3*CCH);
  cvt_f32_f16_kernel<<<cdiv(4*CCH*CCH,TPB),TPB,0,stream>>>(fc1_w, fc1h, 4*CCH*CCH);
  cvt_f32_f16_kernel<<<cdiv(CCH*4*CCH,TPB),TPB,0,stream>>>(fc2_w, fc2h, CCH*4*CCH);

  // fused ln_cf + mln
  ln_cf_mln_kernel<<<cdiv(BL,TPB),TPB,0,stream>>>(x, ln_w, ln_b, mln_w, mln_b, x2);

  const int scanWaves = (BATCH * DIN * NCH) / 2;       // 19200
  for (int p = 0; p < 3; ++p) {
    gather_kernel<<<cdiv(BL*CCH,TPB),TPB,0,stream>>>(x2, xfh, p);
    // xz = xf @ Win^T : M=16000 K=96 N=384
    gemm_wmma_kernel<<<cdiv((BL/16)*(2*DIN/16),8),TPB,0,stream>>>(
        xfh, Winh, xz, nullptr, nullptr, nullptr,
        BL, 2*DIN/16, CCH, CCH, CCH, 2*DIN, 2*DIN, 0, 0);
    conv_silu_kernel<<<cdiv(BL*DIN,TPB),TPB,0,stream>>>(xz, conv_w, conv_b, xc, xch);
    // x_dbl = xc @ Wx^T : N=38 (3 tiles, guarded), padded stride 48
    gemm_wmma_kernel<<<cdiv((BL/16)*3,8),TPB,0,stream>>>(
        xch, Wxh, xdbl, nullptr, nullptr, nullptr,
        BL, 3, DIN, DIN, DIN, XDBL_LD, DTRANK + 2*DSTATE, 0, 0);
    dt_kernel<<<cdiv(BL*DIN,TPB),TPB,0,stream>>>(xdbl, Wdt, bdt, dtb);
    // chunked selective scan
    scan_p1_kernel<<<cdiv(scanWaves,8),TPB,0,stream>>>(dtb, xc, xdbl, A_log, carA, carH);
    scan_p2_kernel<<<cdiv(BATCH*DIN*DSTATE,TPB),TPB,0,stream>>>(carA, carH, hini);
    scan_p3_kernel<<<cdiv(scanWaves,8),TPB,0,stream>>>(dtb, xc, xdbl, xz, A_log, Dp, hini, ygh);
    // direction output into cat columns [p*96, p*96+96), f16
    gemm_wmma_kernel<<<cdiv((BL/16)*(CCH/16),8),TPB,0,stream>>>(
        ygh, Wouth, cath + p*CCH, nullptr, nullptr, nullptr,
        BL, CCH/16, DIN, DIN, DIN, 3*CCH, CCH, 0, 1);
  }

  // out_res = x + cat @ proj_w^T + proj_b   (channel-last f32)
  gemm_wmma_kernel<<<cdiv((BL/16)*(CCH/16),8),TPB,0,stream>>>(
      cath, projh, ores, proj_b, x, nullptr,
      BL, CCH/16, 3*CCH, 3*CCH, 3*CCH, CCH, CCH, 0, 0);
  // h = LN(out_res)
  ln_cl_kernel<<<cdiv(BL,TPB),TPB,0,stream>>>(ores, ln_w, ln_b, hh);
  // g = gelu(h @ fc1^T + fc1_b)  (f16)
  gemm_wmma_kernel<<<cdiv((BL/16)*(4*CCH/16),8),TPB,0,stream>>>(
      hh, fc1h, gh, fc1_b, nullptr, nullptr,
      BL, 4*CCH/16, CCH, CCH, CCH, 4*CCH, 4*CCH, 1, 1);
  // out = out_res + g @ fc2^T + fc2_b, stored channel-first into d_out
  gemm_wmma_kernel<<<cdiv((BL/16)*(CCH/16),8),TPB,0,stream>>>(
      gh, fc2h, d_out, fc2_b, nullptr, ores,
      BL, CCH/16, 4*CCH, 4*CCH, 4*CCH, CCH, CCH, 0, 2);

  (void)in_sizes; (void)n_in; (void)out_size; (void)ws_size;
}
